// Decoder_68539088109633
// MI455X (gfx1250) — matrix operand order repaired and checked by reading, not compiled, not run on hardware
//
#include <hip/hip_runtime.h>
#include <hip/hip_bf16.h>
#include <math.h>

// Model dims (fixed by the reference)
#define L_  6
#define B_  4
#define S_  512
#define D_  768
#define H_  12
#define F_  3072
#define V_  30522
#define DH_ 64
#define T_  (B_ * S_)   // 2048 token rows

typedef __attribute__((ext_vector_type(16))) __bf16 v16bf;
typedef __attribute__((ext_vector_type(8)))  float  v8f;
typedef int v4i_ __attribute__((vector_size(4 * sizeof(int))));

static __device__ __forceinline__ unsigned short f2bf(float f) {
  // round-to-nearest-even fp32 -> bf16
  unsigned int u = __float_as_uint(f);
  unsigned int r = (u + 0x7FFFu + ((u >> 16) & 1u)) >> 16;
  return (unsigned short)r;
}

static __device__ __forceinline__ float gelu_erf(float x) {
  return 0.5f * x * (1.0f + erff(x * 0.70710678118654752f));
}

// ---------------------------------------------------------------------------
// CDNA5 async global->LDS copy (ASYNCcnt path, CDNA5 ISA).
// Builtin prototype (from hipcc diagnostic): (v4i AS1*, v4i AS3*, int, int).
// Guarded: falls back to a synchronous float4 copy if the builtin is absent.
// ---------------------------------------------------------------------------
#if __has_builtin(__builtin_amdgcn_global_load_async_to_lds_b128)
#define HAVE_ASYNC_LDS 1
#else
#define HAVE_ASYNC_LDS 0
#endif

static __device__ __forceinline__ void cp_async_f4(float* lds_dst,
                                                   const float* gsrc) {
#if HAVE_ASYNC_LDS
  float* g = const_cast<float*>(gsrc);
  __builtin_amdgcn_global_load_async_to_lds_b128(
      (__attribute__((address_space(1))) v4i_*)g,
      (__attribute__((address_space(3))) v4i_*)lds_dst, 0, 0);
#else
  *(float4*)lds_dst = *(const float4*)gsrc;
#endif
}

static __device__ __forceinline__ void wait_async0() {
#if HAVE_ASYNC_LDS
#if __has_builtin(__builtin_amdgcn_s_wait_asynccnt)
  __builtin_amdgcn_s_wait_asynccnt(0);
#else
  asm volatile("s_wait_asynccnt 0x0" ::: "memory");
#endif
#endif
}

// -----------------------------------------------------------------------------
// Tiled GEMM: C[M,N] = act(A[M,K] @ W[K,N] + bias[N])
// BM=128, BN=128, BK=32. 256 threads = 8 wave32 arranged 4(M) x 2(N); each
// wave owns a 32x64 sub-tile -> 2 A-frags x 4 B-frags = 8 v_wmma per k-step
// (2x the WMMA-per-LDS-byte of a 16x32 wave tile). fp32 -> bf16 RNE happens
// while staging tiles into LDS; B is stored transposed ([n][k]) so each
// lane's K-pairs are contiguous 32-bit LDS reads per the ISA 16-bit layouts.
// -----------------------------------------------------------------------------
template <int ACT>
__global__ __launch_bounds__(256) void gemm_bf16_wmma(
    const float* __restrict__ A, const float* __restrict__ W,
    const float* __restrict__ bias, float* __restrict__ C,
    int M, int N, int K)
{
  __shared__ __align__(16) unsigned short As[128 * 32];  // [row][k]
  __shared__ __align__(16) unsigned short Bs[128 * 32];  // [n][k] (transposed)

  const int tid   = threadIdx.x;
  const int lane  = tid & 31;
  const int wv    = tid >> 5;
  const int waveM = wv >> 1;   // 0..3 -> rows waveM*32 .. +31
  const int waveN = wv & 1;    // 0..1 -> cols waveN*64 .. +63
  const int half  = lane >> 4; // 0/1
  const int lr    = lane & 15;
  const int mTile = blockIdx.y * 128;
  const int nTile = blockIdx.x * 128;

  v8f acc[2][4] = {};

  const bool fastB = ((N & 3) == 0);

  for (int k0 = 0; k0 < K; k0 += 32) {
    // L2 prefetch of the next k-tile (global_prefetch_b8)
    if (k0 + 32 < K) {
      __builtin_prefetch(A + (size_t)(mTile + (tid >> 3)) * K + (k0 + 32), 0, 1);
      __builtin_prefetch(W + (size_t)(k0 + 32 + (tid >> 3)) * N + nTile, 0, 1);
    }

    // ---- stage A tile (128 x 32) as bf16; float4 global reads (K % 4 == 0) --
#pragma unroll
    for (int j = 0; j < 4; ++j) {
      int v   = j * 256 + tid;        // 0..1023 float4 units
      int row = v >> 3;               // 8 float4 per 32-wide row
      int kq  = (v & 7) << 2;
      const float4 f =
          *(const float4*)(A + (size_t)(mTile + row) * K + (k0 + kq));
      As[row * 32 + kq + 0] = f2bf(f.x);
      As[row * 32 + kq + 1] = f2bf(f.y);
      As[row * 32 + kq + 2] = f2bf(f.z);
      As[row * 32 + kq + 3] = f2bf(f.w);
    }
    // ---- stage B tile (32 x 128) transposed -> Bs[n][k] ----
    if (fastB && nTile + 128 <= N) {
#pragma unroll
      for (int j = 0; j < 4; ++j) {
        int v  = j * 256 + tid;       // 0..1023 float4 units
        int kk = v >> 5;              // 32 float4 per 128-wide row
        int nq = (v & 31) << 2;
        const float4 f =
            *(const float4*)(W + (size_t)(k0 + kk) * N + (nTile + nq));
        Bs[(nq + 0) * 32 + kk] = f2bf(f.x);
        Bs[(nq + 1) * 32 + kk] = f2bf(f.y);
        Bs[(nq + 2) * 32 + kk] = f2bf(f.z);
        Bs[(nq + 3) * 32 + kk] = f2bf(f.w);
      }
    } else {
#pragma unroll
      for (int j = 0; j < 16; ++j) {
        int e  = j * 256 + tid;       // 0..4095
        int kk = e >> 7;              // 0..31
        int nn = e & 127;             // 0..127
        int ng = nTile + nn;
        float f = (ng < N) ? W[(size_t)(k0 + kk) * N + ng] : 0.0f;
        Bs[nn * 32 + kk] = f2bf(f);
      }
    }
    __syncthreads();

    // ---- build A fragments (16x32 bf16 per ISA layout) ----
    union { v16bf v; unsigned int u[8]; } af[2];
#pragma unroll
    for (int mi = 0; mi < 2; ++mi) {
      const unsigned int* As32 = (const unsigned int*)As;
      int row = waveM * 32 + mi * 16 + lr;
#pragma unroll
      for (int v = 0; v < 8; ++v) {
        // lanes 0-15: VGPR0-3 -> K=0..7, VGPR4-7 -> K=16..23; lanes 16-31: +8
        int kk = ((v < 4) ? (2 * v) : (2 * v + 8)) + 8 * half;
        af[mi].u[v] = As32[(row * 32 + kk) >> 1];
      }
    }
    // ---- build B fragments (32x16: lanes 0-15 K=0..7,16..23; lanes 16-31 K=8..15,24..31)
    union { v16bf v; unsigned int u[8]; } bf[4];
#pragma unroll
    for (int t = 0; t < 4; ++t) {
      const unsigned int* Bs32 = (const unsigned int*)Bs;
      int c = waveN * 64 + t * 16 + lr;
#pragma unroll
      for (int v = 0; v < 8; ++v) {
        int kk = ((v < 4) ? (2 * v) : (2 * v + 8)) + 8 * half;
        bf[t].u[v] = Bs32[(c * 32 + kk) >> 1];
      }
    }

#pragma unroll
    for (int mi = 0; mi < 2; ++mi)
#pragma unroll
      for (int t = 0; t < 4; ++t)
        acc[mi][t] = __builtin_amdgcn_wmma_f32_16x16x32_bf16(
            false, af[mi].v, false, bf[t].v, (short)0, acc[mi][t], false,
            false);
    __syncthreads();
  }

  // ---- epilogue: bias (+ optional erf-gelu), bounds-checked store ----
  // D layout: VGPR r, lanes 0-15 -> (M=r, N=lane); lanes 16-31 -> (M=8+r)
#pragma unroll
  for (int mi = 0; mi < 2; ++mi) {
#pragma unroll
    for (int t = 0; t < 4; ++t) {
      int col = nTile + waveN * 64 + t * 16 + lr;
      if (col < N) {
        float bv = bias[col];
#pragma unroll
        for (int r = 0; r < 8; ++r) {
          int row = mTile + waveM * 32 + mi * 16 + r + 8 * half;
          if (row < M) {
            float vv = acc[mi][t][r] + bv;
            if (ACT) vv = gelu_erf(vv);
            C[(size_t)row * N + col] = vv;
          }
        }
      }
    }
  }
}

// -----------------------------------------------------------------------------
// Flash-style attention with async-staged K/V tiles.
// Block = 8 wave32, one query row per wave (DH=64 -> 2 dims/lane). K/V are
// staged 64 keys at a time into LDS via GLOBAL_LOAD_ASYNC_TO_LDS_B128
// (ASYNCcnt + s_wait_asynccnt), shared by all 8 query-waves -> 8x less K/V
// global traffic. Online softmax, shuffle-tree dot reduction.
// -----------------------------------------------------------------------------
__global__ __launch_bounds__(256) void attn_kernel(
    const float* __restrict__ Q, const float* __restrict__ Km,
    const float* __restrict__ Vm, float* __restrict__ O, int causal)
{
  __shared__ __align__(16) float Ks[64 * DH_];
  __shared__ __align__(16) float Vs[64 * DH_];

  const int tid  = threadIdx.x;
  const int lane = tid & 31;
  const int wv   = tid >> 5;
  const int qi   = blockIdx.x * 8 + wv;
  const int h    = blockIdx.y;
  const int b    = blockIdx.z;

  const size_t rowQ = (size_t)(b * S_ + qi) * D_ + h * DH_;
  const int d0 = lane * 2;

  const float scale = 0.125f;  // 1/sqrt(64)
  float q0 = Q[rowQ + d0] * scale;
  float q1 = Q[rowQ + d0 + 1] * scale;

  float m = -INFINITY, ssum = 0.0f, o0 = 0.0f, o1 = 0.0f;
  const int qmaxb = blockIdx.x * 8 + 7;

  for (int k0 = 0; k0 < S_; k0 += 64) {
    if (causal && k0 > qmaxb) break;   // uniform over the block

    // ---- stage 64 keys + 64 values (16KB each) into LDS, async ----
#pragma unroll
    for (int j = 0; j < 4; ++j) {
      int v   = j * 256 + tid;        // 0..1023 float4 units
      int row = v >> 4;               // 16 float4 per 64-wide row
      int cq  = (v & 15) << 2;
      size_t g = (size_t)(b * S_ + k0 + row) * D_ + h * DH_ + cq;
      cp_async_f4(&Ks[row * DH_ + cq], Km + g);
      cp_async_f4(&Vs[row * DH_ + cq], Vm + g);
    }
    wait_async0();
    __syncthreads();

    int lim = causal ? (qi - k0 + 1) : 64;
    if (lim > 64) lim = 64;
    for (int kk = 0; kk < lim; ++kk) {
      float s = q0 * Ks[kk * DH_ + d0] + q1 * Ks[kk * DH_ + d0 + 1];
#pragma unroll
      for (int off = 16; off > 0; off >>= 1)
        s += __shfl_xor(s, off, 32);

      float nm = fmaxf(m, s);
      float c  = __expf(m - nm);
      float p  = __expf(s - nm);
      ssum = ssum * c + p;
      o0   = o0 * c + p * Vs[kk * DH_ + d0];
      o1   = o1 * c + p * Vs[kk * DH_ + d0 + 1];
      m = nm;
    }
    __syncthreads();
  }
  float inv = 1.0f / ssum;
  O[rowQ + d0]     = o0 * inv;
  O[rowQ + d0 + 1] = o1 * inv;
}

// -----------------------------------------------------------------------------
// LayerNorm over D=768: out = (x (+resid) - mean) * rstd * g + b.
// One row per block, 256 threads, 3 elements/thread cached in registers so
// out may alias resid (in-place residual LN).
// -----------------------------------------------------------------------------
__global__ __launch_bounds__(256) void ln_kernel(
    float* __restrict__ out, const float* __restrict__ x,
    const float* __restrict__ resid,
    const float* __restrict__ g, const float* __restrict__ bb)
{
  __shared__ float red[256];
  const int row = blockIdx.x;
  const int tid = threadIdx.x;

  float vals[3];
  float s = 0.0f;
#pragma unroll
  for (int j = 0; j < 3; ++j) {
    int idx = tid + j * 256;
    float v = x[(size_t)row * D_ + idx];
    if (resid) v += resid[(size_t)row * D_ + idx];
    vals[j] = v;
    s += v;
  }
  red[tid] = s;
  __syncthreads();
  for (int o = 128; o > 0; o >>= 1) {
    if (tid < o) red[tid] += red[tid + o];
    __syncthreads();
  }
  float mean = red[0] * (1.0f / (float)D_);
  __syncthreads();

  float s2 = 0.0f;
#pragma unroll
  for (int j = 0; j < 3; ++j) {
    float d = vals[j] - mean;
    s2 += d * d;
  }
  red[tid] = s2;
  __syncthreads();
  for (int o = 128; o > 0; o >>= 1) {
    if (tid < o) red[tid] += red[tid + o];
    __syncthreads();
  }
  float rstd = rsqrtf(red[0] * (1.0f / (float)D_) + 1e-12f);

#pragma unroll
  for (int j = 0; j < 3; ++j) {
    int idx = tid + j * 256;
    out[(size_t)row * D_ + idx] = (vals[j] - mean) * rstd * g[idx] + bb[idx];
  }
}

// -----------------------------------------------------------------------------
// Embeddings: x[tok] = word_emb[ids[tok]] + pos_emb[tok % S] + tok_emb[0]
// -----------------------------------------------------------------------------
__global__ __launch_bounds__(256) void embed_kernel(
    const int* __restrict__ ids, const float* __restrict__ we,
    const float* __restrict__ pe, const float* __restrict__ te,
    float* __restrict__ x)
{
  const int tok = blockIdx.x;
  const int s   = tok & (S_ - 1);
  const int id  = ids[tok];
  const int tid = threadIdx.x;
#pragma unroll
  for (int j = 0; j < 3; ++j) {
    int d = tid + j * 256;
    x[(size_t)tok * D_ + d] =
        we[(size_t)id * D_ + d] + pe[(size_t)s * D_ + d] + te[d];
  }
}

// -----------------------------------------------------------------------------
// Host orchestration
// -----------------------------------------------------------------------------
extern "C" void kernel_launch(void* const* d_in, const int* in_sizes, int n_in,
                              void* d_out, int out_size, void* d_ws,
                              size_t ws_size, hipStream_t stream)
{
  (void)in_sizes; (void)n_in; (void)out_size;

  const int*   ids   = (const int*)  d_in[0];
  const float* enc   = (const float*)d_in[1];
  const float* we    = (const float*)d_in[2];
  const float* pe    = (const float*)d_in[3];
  const float* te    = (const float*)d_in[4];
  const float* emb_g = (const float*)d_in[5];
  const float* emb_b = (const float*)d_in[6];
  const float* Wq    = (const float*)d_in[7];
  const float* bq    = (const float*)d_in[8];
  const float* Wk    = (const float*)d_in[9];
  const float* bk    = (const float*)d_in[10];
  const float* Wv    = (const float*)d_in[11];
  const float* bv    = (const float*)d_in[12];
  const float* Wo    = (const float*)d_in[13];
  const float* bo    = (const float*)d_in[14];
  const float* ln1g  = (const float*)d_in[15];
  const float* ln1b  = (const float*)d_in[16];
  const float* cWq   = (const float*)d_in[17];
  const float* cbq   = (const float*)d_in[18];
  const float* cWk   = (const float*)d_in[19];
  const float* cbk   = (const float*)d_in[20];
  const float* cWv   = (const float*)d_in[21];
  const float* cbv   = (const float*)d_in[22];
  const float* cWo   = (const float*)d_in[23];
  const float* cbo   = (const float*)d_in[24];
  const float* ln2g  = (const float*)d_in[25];
  const float* ln2b  = (const float*)d_in[26];
  const float* Wi    = (const float*)d_in[27];
  const float* bi    = (const float*)d_in[28];
  const float* Wf    = (const float*)d_in[29];
  const float* bf    = (const float*)d_in[30];
  const float* ln3g  = (const float*)d_in[31];
  const float* ln3b  = (const float*)d_in[32];
  const float* Wt    = (const float*)d_in[33];
  const float* bt    = (const float*)d_in[34];
  const float* lnhg  = (const float*)d_in[35];
  const float* lnhb  = (const float*)d_in[36];
  const float* Wdec  = (const float*)d_in[37];
  const float* bdec  = (const float*)d_in[38];
  float* out = (float*)d_out;

  // Workspace: h + 4 token-D buffers + one token-F buffer.
  const size_t nTD = (size_t)T_ * D_;   // 1.57M floats
  const size_t nTF = (size_t)T_ * F_;   // 6.29M floats
  float* h  = (float*)d_ws;
  float* t2 = h  + nTD;
  float* t3 = t2 + nTD;
  float* t4 = t3 + nTD;
  float* t5 = t4 + nTD;
  float* t1;
  if (ws_size >= (5 * nTD + nTF) * sizeof(float)) {
    t1 = t5 + nTD;            // everything fits in ws
  } else {
    t1 = out;                 // deterministic fallback: out (T*V floats) is
                              // rewritten in full by the final GEMM
  }

  const dim3 blk(256);
  const dim3 attnGrid(S_ / 8, H_, B_);

  auto gemm = [&](const float* A, const float* W, const float* bias, float* C,
                  int M, int N, int K, bool act) {
    dim3 grid((unsigned)((N + 127) / 128), (unsigned)((M + 127) / 128));
    if (act)
      gemm_bf16_wmma<1><<<grid, blk, 0, stream>>>(A, W, bias, C, M, N, K);
    else
      gemm_bf16_wmma<0><<<grid, blk, 0, stream>>>(A, W, bias, C, M, N, K);
  };

  // Embeddings + LN
  embed_kernel<<<T_, blk, 0, stream>>>(ids, we, pe, te, t5);
  ln_kernel<<<T_, blk, 0, stream>>>(h, t5, nullptr, emb_g, emb_b);

  for (int l = 0; l < L_; ++l) {
    const size_t wOff  = (size_t)l * D_ * D_;
    const size_t bOff  = (size_t)l * D_;
    const size_t wiOff = (size_t)l * D_ * F_;
    const size_t biOff = (size_t)l * F_;

    // --- causal self-attention + residual LN ---
    gemm(h, Wq + wOff, bq + bOff, t1, T_, D_, D_, false);
    gemm(h, Wk + wOff, bk + bOff, t2, T_, D_, D_, false);
    gemm(h, Wv + wOff, bv + bOff, t3, T_, D_, D_, false);
    attn_kernel<<<attnGrid, blk, 0, stream>>>(t1, t2, t3, t4, 1);
    gemm(t4, Wo + wOff, bo + bOff, t5, T_, D_, D_, false);
    ln_kernel<<<T_, blk, 0, stream>>>(h, t5, h, ln1g + bOff, ln1b + bOff);

    // --- cross-attention over encoder_hidden + residual LN ---
    gemm(h,   cWq + wOff, cbq + bOff, t1, T_, D_, D_, false);
    gemm(enc, cWk + wOff, cbk + bOff, t2, T_, D_, D_, false);
    gemm(enc, cWv + wOff, cbv + bOff, t3, T_, D_, D_, false);
    attn_kernel<<<attnGrid, blk, 0, stream>>>(t1, t2, t3, t4, 0);
    gemm(t4, cWo + wOff, cbo + bOff, t5, T_, D_, D_, false);
    ln_kernel<<<T_, blk, 0, stream>>>(h, t5, h, ln2g + bOff, ln2b + bOff);

    // --- FFN (erf-gelu) + residual LN ---
    gemm(h,  Wi + wiOff, bi + biOff, t1, T_, F_, D_, true);
    gemm(t1, Wf + wiOff, bf + bOff,  t5, T_, D_, F_, false);
    ln_kernel<<<T_, blk, 0, stream>>>(h, t5, h, ln3g + bOff, ln3b + bOff);
  }

  // MLM head: transform (dense + gelu + LN), then vocab projection.
  gemm(h, Wt, bt, t5, T_, D_, D_, true);
  ln_kernel<<<T_, blk, 0, stream>>>(t2, t5, nullptr, lnhg, lnhb);
  gemm(t2, Wdec, bdec, out, T_, V_, D_, false);
}
